// COINBlock_44882408243588
// MI455X (gfx1250) — matrix operand order repaired and checked by reading, compiled, not run on hardware
//
#include <hip/hip_runtime.h>

// ---------------------------------------------------------------------------
// COIN block forward-parallel for MI455X (gfx1250, wave32, WMMA).
//   Q = X Wq (rows scaled gamma^t), K = X Wk (rows scaled gamma^-s), V = X Wv;
//   A = mask(Q K^T)  [decay folded into Q/K row scaling];  Out = A V written
//   transposed (Out^T = V^T A^T) so d_out matches the reference [B,I,T] flat
//   layout with coalesced stores.  S_n passthrough to d_out tail.
// bf16 WMMA GEMMs, 128x128x32 tiles, 8 wave32 waves/block, double-buffered
// LDS with batched b128 staging loads (MLP in the feed path).
// ---------------------------------------------------------------------------

typedef __attribute__((ext_vector_type(16))) __bf16 v16bf;
typedef __attribute__((ext_vector_type(8)))  __bf16 v8bf;
typedef __attribute__((ext_vector_type(4)))  __bf16 v4bf;
typedef __attribute__((ext_vector_type(2)))  __bf16 v2bf;
typedef __attribute__((ext_vector_type(8)))  float  v8f;
typedef __attribute__((ext_vector_type(4)))  float  f4;

#define BM 128
#define BN 128
#define BK 32
#define LDST 40          // padded LDS row stride (80B: 16B-aligned rows, skewed banks)
#define NTHREADS 256     // 8 wave32 waves: 4 (M) x 2 (N); each wave 32x64
#define GAMMA_L2 (-0.014499569695115089f)  // log2(0.99)

__device__ __forceinline__ v4bf cvt4(f4 x) {
  v4bf h;
  h.x = (__bf16)x.x; h.y = (__bf16)x.y; h.z = (__bf16)x.z; h.w = (__bf16)x.w;
  return h;
}

// A-matrix 16x32 bf16 fragment: lane m = lane&15, K = (lane>>4)*16 + i,
// 16 contiguous bf16 -> chunks at +0 and +8.
__device__ __forceinline__ v16bf frag_a(const __bf16* p) {
  v8bf lo = *(const v8bf*)(p);
  v8bf hi = *(const v8bf*)(p + 8);
  return __builtin_shufflevector(lo, hi, 0,1,2,3,4,5,6,7,8,9,10,11,12,13,14,15);
}
// B-matrix 32x16 bf16 fragment: lane n = lane&15, K = (lane>>4)*16 + i,
// 16 contiguous bf16 -> chunks at +0 and +8.
__device__ __forceinline__ v16bf frag_b(const __bf16* p) {
  v8bf lo = *(const v8bf*)(p);
  v8bf hi = *(const v8bf*)(p + 8);
  return __builtin_shufflevector(lo, hi, 0,1,2,3,4,5,6,7,8,9,10,11,12,13,14,15);
}

// ---------------------------------------------------------------------------
// Kernel 1: Y[row,n] = bf16( rowscale(row) * sum_k X[row,k] * W[k,n] )
//   which 0: Wq, gamma^t ; 1: Wk, gamma^-t ; 2: Wv, 1.0
// ---------------------------------------------------------------------------
__global__ __launch_bounds__(NTHREADS)
void qkv_gemm_bf16(const float* __restrict__ X,
                   const float* __restrict__ Wq,
                   const float* __restrict__ Wk,
                   const float* __restrict__ Wv,
                   __bf16* __restrict__ Qb,
                   __bf16* __restrict__ Kb,
                   __bf16* __restrict__ Vb) {
  __shared__ __attribute__((aligned(16))) __bf16 lA[2][BM][LDST];
  __shared__ __attribute__((aligned(16))) __bf16 lB[2][BN][LDST];
  const int which = blockIdx.z;
  const float* __restrict__ W = (which == 0) ? Wq : (which == 1) ? Wk : Wv;
  __bf16* __restrict__ Y = (which == 0) ? Qb : (which == 1) ? Kb : Vb;

  const int KD = 1024;
  const int m0 = blockIdx.y * BM;
  const int n0 = blockIdx.x * BN;
  const int lane = threadIdx.x & 31;
  const int wv   = threadIdx.x >> 5;
  const int wm   = (wv & 3) * 32;
  const int wn   = (wv >> 2) * 64;
  const int ml   = lane & 15;
  const int kh   = lane >> 4;

  // Stage 128x32 fp32->bf16 A tile + 32x128 transposed W tile into buffer p.
  // Phase 1: batch ALL global b128 loads into registers (MLP), phase 2: cvt+store.
  auto stage = [&](int k0, int p) {
    f4 xa[4], wa[2], wb[2];
    #pragma unroll
    for (int i = 0; i < 4; ++i) {                 // 1024 float4 chunks of X tile
      int e4 = threadIdx.x + i * NTHREADS;
      int r  = e4 >> 3;
      int k4 = (e4 & 7) << 2;
      xa[i] = *(const f4*)&X[(size_t)(m0 + r) * KD + (k0 + k4)];
    }
    #pragma unroll
    for (int i = 0; i < 2; ++i) {                 // 512 k-paired f4 chunks of W
      int e  = threadIdx.x + i * NTHREADS;
      int kp = e >> 5;                            // k-pair 0..15
      int n4 = (e & 31) << 2;
      wa[i] = *(const f4*)&W[(size_t)(k0 + 2 * kp)     * 1024 + (n0 + n4)];
      wb[i] = *(const f4*)&W[(size_t)(k0 + 2 * kp + 1) * 1024 + (n0 + n4)];
    }
    #pragma unroll
    for (int i = 0; i < 4; ++i) {
      int e4 = threadIdx.x + i * NTHREADS;
      int r  = e4 >> 3;
      int k4 = (e4 & 7) << 2;
      *(v4bf*)&lA[p][r][k4] = cvt4(xa[i]);
    }
    #pragma unroll
    for (int i = 0; i < 2; ++i) {
      int e  = threadIdx.x + i * NTHREADS;
      int kp = e >> 5;
      int n4 = (e & 31) << 2;
      v4bf ha = cvt4(wa[i]), hb = cvt4(wb[i]);
      #pragma unroll
      for (int j = 0; j < 4; ++j) {               // (k,k+1) contiguous: b32 store
        v2bf pr; pr.x = ha[j]; pr.y = hb[j];
        *(v2bf*)&lB[p][n4 + j][2 * kp] = pr;
      }
    }
  };

  const v8f vz = {0.f,0.f,0.f,0.f,0.f,0.f,0.f,0.f};
  v8f acc[2][4];
  #pragma unroll
  for (int mi = 0; mi < 2; ++mi)
    #pragma unroll
    for (int ni = 0; ni < 4; ++ni) acc[mi][ni] = vz;

  stage(0, 0);
  int p = 0;
  for (int k0 = 0; k0 < KD; k0 += BK) {
    __syncthreads();
    if (k0 + 2 * BK < KD)  // global_prefetch_b8 two tiles ahead, near caches
      __builtin_prefetch(&X[(size_t)(m0 + (threadIdx.x >> 1)) * KD + k0 + 2 * BK], 0, 3);
    if (k0 + BK < KD) stage(k0 + BK, p ^ 1);
    v16bf af[2], bfv[4];
    #pragma unroll
    for (int mi = 0; mi < 2; ++mi) af[mi]  = frag_a(&lA[p][wm + mi * 16 + ml][kh * 16]);
    #pragma unroll
    for (int ni = 0; ni < 4; ++ni) bfv[ni] = frag_b(&lB[p][wn + ni * 16 + ml][kh * 16]);
    #pragma unroll
    for (int mi = 0; mi < 2; ++mi)
      #pragma unroll
      for (int ni = 0; ni < 4; ++ni)
        acc[mi][ni] = __builtin_amdgcn_wmma_f32_16x16x32_bf16(
            false, af[mi], false, bfv[ni], (short)0, acc[mi][ni], false, false);
    p ^= 1;
  }

  #pragma unroll
  for (int mi = 0; mi < 2; ++mi) {
    #pragma unroll
    for (int r = 0; r < 8; ++r) {
      const int row = m0 + wm + mi * 16 + kh * 8 + r;
      const int t = row & 4095;  // row = b*T + t
      float rs = 1.0f;
      if (which == 0)      rs = exp2f(GAMMA_L2 * (float)t);    // gamma^t
      else if (which == 1) rs = exp2f(-GAMMA_L2 * (float)t);   // gamma^-t
      #pragma unroll
      for (int ni = 0; ni < 4; ++ni) {
        const int col = n0 + wn + ni * 16 + ml;
        Y[(size_t)row * 1024 + col] = (__bf16)(acc[mi][ni][r] * rs);
      }
    }
  }
}

// ---------------------------------------------------------------------------
// Kernel 2: A[b,t,s] = (s<=t) ? sum_c Qb[b,t,c]*Kb[b,s,c] : 0   (bf16 out)
// Fully-above-diagonal tiles skipped (never read downstream).
// ---------------------------------------------------------------------------
__global__ __launch_bounds__(NTHREADS)
void scores_gemm(const __bf16* __restrict__ Qb,
                 const __bf16* __restrict__ Kb,
                 __bf16* __restrict__ Aw) {
  const int T = 4096, C = 1024;
  const int b  = blockIdx.z;
  const int t0 = blockIdx.y * BM;
  const int s0 = blockIdx.x * BN;
  if (s0 > t0 + (BM - 1)) return;

  __shared__ __attribute__((aligned(16))) __bf16 lA[2][BM][LDST];
  __shared__ __attribute__((aligned(16))) __bf16 lB[2][BN][LDST];
  const int lane = threadIdx.x & 31;
  const int wv   = threadIdx.x >> 5;
  const int wm   = (wv & 3) * 32;
  const int wn   = (wv >> 2) * 64;
  const int ml   = lane & 15;
  const int kh   = lane >> 4;

  // both operands are row-major bf16 rows: pure 16B copies, loads batched
  auto stage = [&](int k0, int p) {
    v8bf qa[2], ka[2];
    #pragma unroll
    for (int i = 0; i < 2; ++i) {
      int e8 = threadIdx.x + i * NTHREADS;
      int r  = e8 >> 2;
      int k8 = (e8 & 3) << 3;
      qa[i] = *(const v8bf*)&Qb[((size_t)b * T + t0 + r) * C + (k0 + k8)];
      ka[i] = *(const v8bf*)&Kb[((size_t)b * T + s0 + r) * C + (k0 + k8)];
    }
    #pragma unroll
    for (int i = 0; i < 2; ++i) {
      int e8 = threadIdx.x + i * NTHREADS;
      int r  = e8 >> 2;
      int k8 = (e8 & 3) << 3;
      *(v8bf*)&lA[p][r][k8] = qa[i];
      *(v8bf*)&lB[p][r][k8] = ka[i];
    }
  };

  const v8f vz = {0.f,0.f,0.f,0.f,0.f,0.f,0.f,0.f};
  v8f acc[2][4];
  #pragma unroll
  for (int mi = 0; mi < 2; ++mi)
    #pragma unroll
    for (int ni = 0; ni < 4; ++ni) acc[mi][ni] = vz;

  stage(0, 0);
  int p = 0;
  for (int k0 = 0; k0 < C; k0 += BK) {
    __syncthreads();
    if (k0 + 2 * BK < C)
      __builtin_prefetch(&Qb[((size_t)b * T + t0 + (threadIdx.x >> 1)) * C + k0 + 2 * BK], 0, 3);
    if (k0 + BK < C) stage(k0 + BK, p ^ 1);
    v16bf af[2], bfv[4];
    #pragma unroll
    for (int mi = 0; mi < 2; ++mi) af[mi]  = frag_a(&lA[p][wm + mi * 16 + ml][kh * 16]);
    #pragma unroll
    for (int ni = 0; ni < 4; ++ni) bfv[ni] = frag_b(&lB[p][wn + ni * 16 + ml][kh * 16]);
    #pragma unroll
    for (int mi = 0; mi < 2; ++mi)
      #pragma unroll
      for (int ni = 0; ni < 4; ++ni)
        acc[mi][ni] = __builtin_amdgcn_wmma_f32_16x16x32_bf16(
            false, af[mi], false, bfv[ni], (short)0, acc[mi][ni], false, false);
    p ^= 1;
  }

  #pragma unroll
  for (int mi = 0; mi < 2; ++mi)
    #pragma unroll
    for (int r = 0; r < 8; ++r) {
      const int t = t0 + wm + mi * 16 + kh * 8 + r;
      #pragma unroll
      for (int ni = 0; ni < 4; ++ni) {
        const int s = s0 + wn + ni * 16 + ml;
        const float v = (s <= t) ? acc[mi][ni][r] : 0.0f;
        Aw[((size_t)b * T + t) * T + s] = (__bf16)v;
      }
    }
}

// ---------------------------------------------------------------------------
// Kernel 3: Out^T[b,i,t] = sum_s V[b,s,i] * A[b,t,s]   (fp32, coalesced store)
// Causality bounds the s-loop at t0+BN.
// ---------------------------------------------------------------------------
__global__ __launch_bounds__(NTHREADS)
void out_gemm(const __bf16* __restrict__ Aw,
              const __bf16* __restrict__ Vb,
              float* __restrict__ Out) {
  const int T = 4096, I = 1024;
  const int b  = blockIdx.z;
  const int t0 = blockIdx.x * BN;  // N dim = t
  const int i0 = blockIdx.y * BM;  // M dim = i
  const int kmax = t0 + BN;

  __shared__ __attribute__((aligned(16))) __bf16 lA[2][BM][LDST];
  __shared__ __attribute__((aligned(16))) __bf16 lB[2][BN][LDST];
  const int lane = threadIdx.x & 31;
  const int wv   = threadIdx.x >> 5;
  const int wm   = (wv & 3) * 32;
  const int wn   = (wv >> 2) * 64;
  const int ml   = lane & 15;
  const int kh   = lane >> 4;

  auto stage = [&](int k0, int p) {
    // A operand = V^T: read two adjacent s-rows coalesced, pair (s,s+1) into
    // b32 transposed stores. One s-pair x 8 i-values per thread covers 32x128.
    const int kkp = threadIdx.x >> 4;             // s-pair 0..15
    const int i8  = (threadIdx.x & 15) << 3;      // i chunk
    v8bf va = *(const v8bf*)&Vb[((size_t)b * T + k0 + 2 * kkp)     * I + (i0 + i8)];
    v8bf vb = *(const v8bf*)&Vb[((size_t)b * T + k0 + 2 * kkp + 1) * I + (i0 + i8)];
    // B operand = A^T -> stage A rows directly (16B copies), loads batched
    v8bf ba[2];
    #pragma unroll
    for (int i = 0; i < 2; ++i) {
      int e8 = threadIdx.x + i * NTHREADS;
      int n  = e8 >> 2;
      int k8 = (e8 & 3) << 3;
      ba[i] = *(const v8bf*)&Aw[((size_t)b * T + t0 + n) * T + (k0 + k8)];
    }
    #pragma unroll
    for (int j = 0; j < 8; ++j) {
      v2bf pr; pr.x = va[j]; pr.y = vb[j];
      *(v2bf*)&lA[p][i8 + j][2 * kkp] = pr;
    }
    #pragma unroll
    for (int i = 0; i < 2; ++i) {
      int e8 = threadIdx.x + i * NTHREADS;
      int n  = e8 >> 2;
      int k8 = (e8 & 3) << 3;
      *(v8bf*)&lB[p][n][k8] = ba[i];
    }
  };

  const v8f vz = {0.f,0.f,0.f,0.f,0.f,0.f,0.f,0.f};
  v8f acc[2][4];
  #pragma unroll
  for (int mi = 0; mi < 2; ++mi)
    #pragma unroll
    for (int ni = 0; ni < 4; ++ni) acc[mi][ni] = vz;

  stage(0, 0);
  int p = 0;
  for (int k0 = 0; k0 < kmax; k0 += BK) {
    __syncthreads();
    if (k0 + 2 * BK < kmax)
      __builtin_prefetch(&Aw[((size_t)b * T + t0 + (threadIdx.x >> 1)) * T + k0 + 2 * BK], 0, 3);
    if (k0 + BK < kmax) stage(k0 + BK, p ^ 1);
    v16bf af[2], bfv[4];
    #pragma unroll
    for (int mi = 0; mi < 2; ++mi) af[mi]  = frag_a(&lA[p][wm + mi * 16 + ml][kh * 16]);
    #pragma unroll
    for (int ni = 0; ni < 4; ++ni) bfv[ni] = frag_b(&lB[p][wn + ni * 16 + ml][kh * 16]);
    #pragma unroll
    for (int mi = 0; mi < 2; ++mi)
      #pragma unroll
      for (int ni = 0; ni < 4; ++ni)
        acc[mi][ni] = __builtin_amdgcn_wmma_f32_16x16x32_bf16(
            false, af[mi], false, bfv[ni], (short)0, acc[mi][ni], false, false);
    p ^= 1;
  }

  // Out flat layout per reference: d_out[b*I*T + i*T + t]
  #pragma unroll
  for (int mi = 0; mi < 2; ++mi)
    #pragma unroll
    for (int r = 0; r < 8; ++r) {
      const int i = i0 + wm + mi * 16 + kh * 8 + r;
      #pragma unroll
      for (int ni = 0; ni < 4; ++ni) {
        const int t = t0 + wn + ni * 16 + ml;
        Out[(size_t)b * I * T + (size_t)i * T + t] = acc[mi][ni][r];
      }
    }
}

// ---------------------------------------------------------------------------
extern "C" void kernel_launch(void* const* d_in, const int* in_sizes, int n_in,
                              void* d_out, int out_size, void* d_ws, size_t ws_size,
                              hipStream_t stream) {
  (void)in_sizes; (void)n_in; (void)out_size; (void)ws_size;
  const float* X  = (const float*)d_in[0];
  // d_in[1] = att_mask (unused by reference path)
  const float* Sn = (const float*)d_in[2];
  const float* Wq = (const float*)d_in[3];
  const float* Wk = (const float*)d_in[4];
  const float* Wv = (const float*)d_in[5];
  float* out = (float*)d_out;

  const int B = 2, T = 4096, I = 1024, C = 1024;

  // workspace: Q,K,V bf16 (16MB each) + A bf16 (64MB) = ~112MB
  char* ws = (char*)d_ws;
  const size_t qkvBytes = (size_t)B * T * C * sizeof(__bf16);
  __bf16* Qb = (__bf16*)(ws);
  __bf16* Kb = (__bf16*)(ws + qkvBytes);
  __bf16* Vb = (__bf16*)(ws + 2 * qkvBytes);
  __bf16* Aw = (__bf16*)(ws + 3 * qkvBytes);

  dim3 blk(NTHREADS);
  qkv_gemm_bf16<<<dim3(C / BN, (B * T) / BM, 3), blk, 0, stream>>>(
      X, Wq, Wk, Wv, Qb, Kb, Vb);
  scores_gemm<<<dim3(T / BN, T / BM, B), blk, 0, stream>>>(Qb, Kb, Aw);
  out_gemm<<<dim3(T / BN, I / BM, B), blk, 0, stream>>>(Aw, Vb, out);

  // second tuple output: S_n passthrough into the tail of d_out
  hipMemcpyAsync(out + (size_t)B * T * C, Sn, (size_t)B * C * C * sizeof(float),
                 hipMemcpyDeviceToDevice, stream);
}
